// AdderVDSR_16209206575287
// MI455X (gfx1250) — matrix operand order repaired and checked by reading, compiled, not run on hardware
//
#include <hip/hip_runtime.h>
#include <hip/hip_bf16.h>

typedef __attribute__((ext_vector_type(16))) _Float16 v16h;
typedef __attribute__((ext_vector_type(8)))  float    v8f;
typedef unsigned int u32x4 __attribute__((ext_vector_type(4)));
typedef int          i32x4 __attribute__((ext_vector_type(4)));
typedef int          i32x8 __attribute__((ext_vector_type(8)));

// ---------------------------------------------------------------------------
// im2col gather: value of input[b, c, h0+ki-1, w0+kj-1] for flattened k=c*9+tap
// ---------------------------------------------------------------------------
__device__ __forceinline__ float im2col_val(const float* __restrict__ in,
                                            int b, int h0, int w0,
                                            int H, int W, int Cin, int k) {
    int c   = k / 9;
    int tap = k - 9 * c;
    int ki  = tap / 3;
    int kj  = tap - 3 * ki;
    int hh  = h0 + ki - 1;
    int ww  = w0 + kj - 1;
    if (c >= Cin || hh < 0 || hh >= H || ww < 0 || ww >= W) return 0.0f;
    return in[(((size_t)b * Cin + c) * H + hh) * W + ww];
}

// ---------------------------------------------------------------------------
// Generic 3x3 conv (stride 1, pad 1) as im2col GEMM on v_wmma_f32_16x16x32_f16.
// One wave owns a 16(M: pixels) x 16(N: out-channels) tile; K = Cin*9, padded
// to multiples of 32 with zeros. f32 accumulation.
// ---------------------------------------------------------------------------
template <bool RELU, bool SHUFFLE, bool RESID>
__global__ void conv3_wmma(const float* __restrict__ in,
                           const float* __restrict__ wf,    // [Cout, Cin*9]
                           const float* __restrict__ bias,  // [Cout]
                           const float* __restrict__ resid,
                           float* __restrict__ out,
                           int B, int Cin, int Cout, int H, int W) {
    const int K      = Cin * 9;
    const int Mtiles = (B * H * W) >> 4;
    const int Ntiles = (Cout + 15) >> 4;

    const int wave = blockIdx.x * (blockDim.x >> 5) + (threadIdx.x >> 5);
    if (wave >= Mtiles * Ntiles) return;  // wave-uniform: EXEC all-ones for WMMA

    const int lane  = threadIdx.x & 31;
    const int group = lane >> 4;
    const int lr    = lane & 15;

    const int mt = wave % Mtiles;
    const int nt = wave / Mtiles;

    const int m  = mt * 16 + lr;
    const int HW = H * W;
    const int b0 = m / HW;
    const int r0 = m - b0 * HW;
    const int h0 = r0 / W;
    const int w0 = r0 - h0 * W;

    const int n = nt * 16 + lr;

    v8f acc = {};
    const int ktiles = (K + 31) >> 5;
    for (int kt = 0; kt < ktiles; ++kt) {
        const int kbase = kt << 5;
        v16h afrag, bfrag;
#pragma unroll
        for (int i = 0; i < 16; ++i) {
            // A (16-bit 16x32): group0 K in {0-7,16-23}, group1 {8-15,24-31}
            const int vg = i >> 1, hi = i & 1;
            const int ka = kbase + ((vg & 3) << 1) + hi + ((vg >> 2) << 4) + (group << 3);
            float av = (ka < K) ? im2col_val(in, b0, h0, w0, H, W, Cin, ka) : 0.0f;
            afrag[i] = (_Float16)av;
            // B (32x16): same K order as A: group0 K in {0-7,16-23}, group1 {8-15,24-31}
            const int kb = kbase + (group << 3) + (i & 7) + ((i >> 3) << 4);
            float bv = (kb < K && n < Cout) ? wf[(size_t)n * K + kb] : 0.0f;
            bfrag[i] = (_Float16)bv;
        }
        acc = __builtin_amdgcn_wmma_f32_16x16x32_f16(
            false, afrag, false, bfrag, (short)0, acc, false, false);
    }

    if (n < Cout) {
        const float bn = bias[n];
#pragma unroll
        for (int v = 0; v < 8; ++v) {
            const int Mrow = v + (group << 3);
            const int mg   = mt * 16 + Mrow;
            const int b2   = mg / HW;
            const int r2   = mg - b2 * HW;
            const int h2   = r2 / W;
            const int w2   = r2 - h2 * W;

            float val = acc[v] + bn;
            if (RELU) val = fmaxf(val, 0.0f);

            size_t addr;
            if (SHUFFLE) {
                const int co = n >> 2;
                const int ri = (n >> 1) & 1;
                const int rj = n & 1;
                addr = (((size_t)b2 * (Cout >> 2) + co) * (2 * H) + (2 * h2 + ri)) * (size_t)(2 * W)
                       + (2 * w2 + rj);
            } else {
                addr = (((size_t)b2 * Cout + n) * H + h2) * (size_t)W + w2;
            }
            if (RESID) val += resid[addr];
            out[addr] = val;
        }
    }
}

// ---------------------------------------------------------------------------
// Weight pre-transpose: adder_w [L=16][O=64][K=576] -> wT [L][K][O]
// Makes each 16-in-channel chunk (144 k-rows x 64 outch) a CONTIGUOUS
// 9216-float region, so the TDM can stage it into LDS as a 1D tile.
// ---------------------------------------------------------------------------
__global__ void transpose_adder_w(const float* __restrict__ w, float* __restrict__ wT,
                                  int total) {
    int idx = blockIdx.x * blockDim.x + threadIdx.x;
    if (idx >= total) return;
    const int o   = idx & 63;
    const int rem = idx >> 6;          // l*576 + k
    const int k   = rem % 576;
    const int l   = rem / 576;
    wT[idx] = w[((size_t)l * 64 + o) * 576 + k];
}

// ---------------------------------------------------------------------------
// TDM descriptor for a contiguous copy of 9216 floats (one weight chunk) into
// LDS. D# layout per CDNA5 ISA:
//   group0: [1:0]=count=1, [63:32]=lds_addr, [120:64]=global_addr, [127:126]=2
//   group1: data_size=2 (4B); tensor_dim0=tile_dim0=9216; tensor_dim1=tile_dim1=1;
//           tensor_dim0_stride=9216; groups 2/3 zero (<=2D tensor).
// ---------------------------------------------------------------------------
#define ADD_D  64
#define ADD_CG 16
#define WCHUNK (ADD_CG * 9 * ADD_D)   // 9216 floats = 36864 B

__device__ __forceinline__ void tdm_load_wchunk(const float* gsrc, unsigned lds_byte_addr) {
    const unsigned long long ga = (unsigned long long)(uintptr_t)gsrc;
    u32x4 g0;
    g0[0] = 1u;                                   // count=1, user mode
    g0[1] = lds_byte_addr;                        // LDS destination
    g0[2] = (unsigned)ga;                         // global_addr[31:0]
    g0[3] = (unsigned)((ga >> 32) & 0x1FFFFFFu)   // global_addr[56:32]
            | (2u << 30);                         // type=2 ("image")
    const unsigned N = WCHUNK;                    // 9216 elements (4B each)
    i32x8 g1;
    g1[0] = (int)(2u << 16);                      // data_size=4B; no mask/pad/iterate
    g1[1] = (int)((N & 0xFFFFu) << 16);           // tensor_dim0[15:0]
    g1[2] = (int)(((N >> 16) & 0xFFFFu) | (1u << 16)); // tensor_dim0[31:16] | tensor_dim1=1
    g1[3] = (int)((N & 0xFFFFu) << 16);           // tensor_dim1[31:16]=0 | tile_dim0
    g1[4] = 1;                                    // tile_dim1=1 | tile_dim2=0
    g1[5] = (int)N;                               // tensor_dim0_stride[31:0]
    g1[6] = 0;                                    // stride0[47:32] | stride1[15:0]
    g1[7] = 0;                                    // stride1[47:16]
    const i32x4 z4 = {0, 0, 0, 0};
#if defined(__clang_major__) && (__clang_major__ >= 23)
    const i32x8 z8 = {0, 0, 0, 0, 0, 0, 0, 0};
    __builtin_amdgcn_tensor_load_to_lds(g0, g1, z4, z4, z8, 0);
#else
    __builtin_amdgcn_tensor_load_to_lds(g0, g1, z4, z4, 0);
#endif
}

// ---------------------------------------------------------------------------
// AdderNet 3x3 layer + ReLU: out = relu(-sum_{c,tap} |patch - w|), 64ch->64ch.
// 16x16 spatial tile / 256-thread block; input channels in chunks of 16.
// Weight slab (9216 floats) staged by the Tensor Data Mover, DOUBLE-BUFFERED
// in LDS so chunk i+1's DMA overlaps chunk i's abs-diff compute.
// LDS: 2 x 36864 B weights + 20736 B halo patch = 94464 B.
// ---------------------------------------------------------------------------
__global__ void __launch_bounds__(256)
adder_relu(const float* __restrict__ in,
           const float* __restrict__ wT,   // this layer, [576][64] (k-major)
           float* __restrict__ out,
           int B, int H, int W) {
    __shared__ float sP[ADD_CG * 18 * 18];
    __shared__ float sW[2][WCHUNK];

    const int tiles_w = W >> 4;
    const int tiles_h = H >> 4;
    const int bt = blockIdx.x;
    const int tw = bt % tiles_w;
    const int th = (bt / tiles_w) % tiles_h;
    const int b  = bt / (tiles_w * tiles_h);

    const int tid = threadIdx.x;
    const int ty  = tid >> 4;
    const int tx  = tid & 15;

    // scalar (SGPR) condition -> real s_cbranch; TDM ignores EXEC, so the
    // issuing block must be branch-skipped (not merely predicated) on waves 1-7
    const bool wave0 = (__builtin_amdgcn_readfirstlane((int)threadIdx.x) < 32);

    if (wave0) tdm_load_wchunk(wT, (unsigned)(uintptr_t)&sW[0][0]);  // chunk 0

    float acc[ADD_D];
#pragma unroll
    for (int o = 0; o < ADD_D; ++o) acc[o] = 0.0f;

    for (int ci = 0; ci < ADD_D / ADD_CG; ++ci) {
        const int c0 = ci * ADD_CG;
        if (wave0) __builtin_amdgcn_s_wait_tensorcnt(0);   // chunk ci weights landed
        __syncthreads();  // TDM data visible; all waves done with previous buffers
        if (wave0 && (ci + 1) < ADD_D / ADD_CG)            // prefetch chunk ci+1 via TDM
            tdm_load_wchunk(wT + (size_t)(c0 + ADD_CG) * 9 * ADD_D,
                            (unsigned)(uintptr_t)&sW[(ci + 1) & 1][0]);

        // stage input halo patch for this channel chunk (VMEM -> LDS)
        for (int idx = tid; idx < ADD_CG * 324; idx += 256) {
            const int c = idx / 324;
            const int r = idx - c * 324;
            const int y = r / 18;
            const int x = r - y * 18;
            const int gh = th * 16 + y - 1;
            const int gw = tw * 16 + x - 1;
            float v = 0.0f;
            if (gh >= 0 && gh < H && gw >= 0 && gw < W) {
                const size_t off = (((size_t)b * ADD_D + (c0 + c)) * H + gh) * W + gw;
                v = in[off];
                if (c0 + ADD_CG < ADD_D)                   // warm L2/GL1 for next chunk
                    __builtin_prefetch(&in[off + (size_t)ADD_CG * H * W], 0, 1);
            }
            sP[idx] = v;
        }
        __syncthreads();

        const float* __restrict__ wbuf = &sW[ci & 1][0];
        for (int c = 0; c < ADD_CG; ++c) {
            float p[9];
#pragma unroll
            for (int ki = 0; ki < 3; ++ki)
#pragma unroll
                for (int kj = 0; kj < 3; ++kj)
                    p[ki * 3 + kj] = sP[c * 324 + (ty + ki) * 18 + (tx + kj)];
#pragma unroll
            for (int tap = 0; tap < 9; ++tap) {
                const float v = p[tap];
                const float4* wr = (const float4*)&wbuf[(c * 9 + tap) * ADD_D];
#pragma unroll
                for (int o4 = 0; o4 < ADD_D / 4; ++o4) {
                    const float4 w4 = wr[o4];
                    acc[o4 * 4 + 0] += fabsf(v - w4.x);
                    acc[o4 * 4 + 1] += fabsf(v - w4.y);
                    acc[o4 * 4 + 2] += fabsf(v - w4.z);
                    acc[o4 * 4 + 3] += fabsf(v - w4.w);
                }
            }
        }
    }

    const int gh = th * 16 + ty;
    const int gw = tw * 16 + tx;
#pragma unroll
    for (int o = 0; o < ADD_D; ++o)
        out[(((size_t)b * ADD_D + o) * H + gh) * W + gw] = fmaxf(0.0f, -acc[o]);
}

// ---------------------------------------------------------------------------
// Orchestration. B=2, C=3, H=W=128, d=64, m=16; upsampled 256x256.
// Workspace: residual [2,3,256,256] | h0,h1 [2,64,256,256] | wT [16,576,64]
// ---------------------------------------------------------------------------
extern "C" void kernel_launch(void* const* d_in, const int* in_sizes, int n_in,
                              void* d_out, int out_size, void* d_ws, size_t ws_size,
                              hipStream_t stream) {
    const float* x       = (const float*)d_in[0];
    const float* up_w    = (const float*)d_in[1];
    const float* up_b    = (const float*)d_in[2];
    const float* in_w    = (const float*)d_in[3];
    const float* in_b    = (const float*)d_in[4];
    const float* adder_w = (const float*)d_in[5];
    const float* out_w   = (const float*)d_in[6];
    const float* out_b   = (const float*)d_in[7];

    const int B = 2, C = 3, H = 128, W = 128, D = 64, NLAYER = 16;
    const int H2 = 2 * H, W2 = 2 * W;

    float* ws       = (float*)d_ws;
    float* residual = ws;                                   // 393216 f
    float* h0       = residual + (size_t)B * C * H2 * W2;   // 8388608 f
    float* h1       = h0 + (size_t)B * D * H2 * W2;         // 8388608 f
    float* wT       = h1 + (size_t)B * D * H2 * W2;         // 589824 f

    // 0) transpose adder weights -> [l][k][o] (contiguous TDM chunks)
    {
        const int total = NLAYER * D * D * 9;               // 589824
        transpose_adder_w<<<(total + 255) / 256, 256, 0, stream>>>(adder_w, wT, total);
    }
    // 1) up conv 3->12 + PixelShuffle(2) + bias -> residual
    {
        int waves = ((B * H * W) >> 4) * 1;                 // 2048
        conv3_wmma<false, true, false><<<(waves + 7) / 8, 256, 0, stream>>>(
            x, up_w, up_b, nullptr, residual, B, C, 4 * C, H, W);
    }
    // 2) in conv 3->64 + ReLU -> h0
    {
        int waves = ((B * H2 * W2) >> 4) * 4;               // 32768
        conv3_wmma<true, false, false><<<(waves + 7) / 8, 256, 0, stream>>>(
            residual, in_w, in_b, nullptr, h0, B, C, D, H2, W2);
    }
    // 3) 16 adder+relu layers, ping-pong h0 <-> h1, TDM-staged weights
    float* cur = h0;
    float* nxt = h1;
    const int adder_blocks = B * (H2 / 16) * (W2 / 16);     // 512
    for (int l = 0; l < NLAYER; ++l) {
        adder_relu<<<adder_blocks, 256, 0, stream>>>(
            cur, wT + (size_t)l * D * D * 9, nxt, B, H2, W2);
        float* t = cur; cur = nxt; nxt = t;
    }
    // 4) out conv 64->3 + bias + global residual -> d_out
    {
        int waves = ((B * H2 * W2) >> 4) * 1;               // 8192
        conv3_wmma<false, false, true><<<(waves + 7) / 8, 256, 0, stream>>>(
            cur, out_w, out_b, residual, (float*)d_out, B, D, C, H2, W2);
    }
}
